// MultiHeadAttention_67001489818028
// MI455X (gfx1250) — matrix operand order repaired and checked by reading, not compiled, not run on hardware
//
#include <hip/hip_runtime.h>

typedef __attribute__((ext_vector_type(16))) _Float16 v16h;
typedef __attribute__((ext_vector_type(8)))  _Float16 v8h;
typedef __attribute__((ext_vector_type(8)))  float    v8f;

#define NB   4
#define CCH  256
#define LSEQ 4096
#define NHD  8
#define HD   32
#define CO3  768
#define QSCALE 0.17677669529663689f   // 1/sqrt(32)

// workspace layout, in _Float16 units
#define WH2_SZ (8*48*32*16)            // 196608 halves: swizzled f16 W_qkv
#define QKV_SZ (NB*NHD*LSEQ*HD)        // 4194304 halves per matrix
#define Q_OFF  (WH2_SZ)
#define K_OFF  (Q_OFF + QKV_SZ)
#define V_OFF  (K_OFF + QKV_SZ)

#define KPAD 40                        // K LDS row stride (halves): 80B, 16B-aligned, conflict-free
#define VPAD 72                        // V LDS row stride (halves): 144B, 16B-aligned, conflict-free

// ---------------------------------------------------------------------------
// CDNA5 async copy: 16 bytes global -> LDS per lane, tracked by ASYNCcnt.
// VFLAT async encoding: VDST = LDS byte address VGPR, VADDR = 64-bit global.
// ---------------------------------------------------------------------------
__device__ __forceinline__ void async_ld16(void* lds, const void* g) {
    asm volatile("global_load_async_to_lds_b128 %0, %1, off"
                 :: "v"((unsigned int)(unsigned long long)lds),
                    "v"((unsigned long long)g)
                 : "memory");
}
__device__ __forceinline__ void wait_async_le2() {
    asm volatile("s_wait_asynccnt 0x2" ::: "memory");
}
__device__ __forceinline__ void wait_async_0() {
    asm volatile("s_wait_asynccnt 0x0" ::: "memory");
}

// ---------------------------------------------------------------------------
// k0: convert W_qkv (256x768 f32, row-major) to f16, pre-swizzled into this file's
// B-fragment order (contiguous k, same as A): frag(kt,nt)[lane][j] = W[kt*32 + j + 16*(lane/16)]
//                                                [nt*16 + lane%16]
// ---------------------------------------------------------------------------
__global__ __launch_bounds__(256) void k0_swizzleW(const float* __restrict__ W,
                                                   _Float16* __restrict__ wh2) {
    int idx  = blockIdx.x * 256 + threadIdx.x;   // < 196608
    int j    = idx & 15;
    int lane = (idx >> 4) & 31;
    int idx2 = idx >> 9;
    int nt   = idx2 % 48;
    int kt   = idx2 / 48;
    int k    = kt * 32 + j + 16 * (lane >> 4);
    int n    = nt * 16 + (lane & 15);
    wh2[idx] = (_Float16)W[k * CO3 + n];
}

// ---------------------------------------------------------------------------
// k1: qkv = (x^T + pe) @ W  via v_wmma_f32_16x16x32_f16.
// Q,K written [n][h][l][32] (Q pre-scaled by 1/sqrt(d)); V written TRANSPOSED
// [n][h][32][L] so k2 can async-copy contiguous V rows straight into LDS.
// Store epilogue is branchless: both layouts are base + lq*stride.
// ---------------------------------------------------------------------------
__global__ __launch_bounds__(256) void k1_qkv(const float* __restrict__ x,
                                              const _Float16* __restrict__ wh2,
                                              _Float16* __restrict__ q,
                                              _Float16* __restrict__ kk,
                                              _Float16* __restrict__ v) {
    __shared__ _Float16 Alds[64 * 264];          // 64 rows x 256 (+8 pad) halves
    const int n     = blockIdx.y;
    const int lbase = blockIdx.x * 64;
    const int tid   = threadIdx.x;

    for (int i = tid; i < 64 * 256; i += 256) {
        int c = i / 64, l = i % 64;
        float xv   = x[(size_t)(n * CCH + c) * LSEQ + lbase + l];
        float e    = (float)(c & ~1) * (1.0f / 256.0f);
        float freq = __powf(10000.0f, -e);
        float ang  = (float)(lbase + l) * freq;
        float pe   = (c & 1) ? __cosf(ang) : __sinf(ang);
        Alds[l * 264 + c] = (_Float16)(xv + pe);
    }
    __syncthreads();

    const int wv = tid >> 5, lane = tid & 31, hi = lane >> 4, ln = lane & 15;

    for (int nt = wv; nt < 48; nt += 8) {
        int mat = nt / 16;                       // 0=Q 1=K 2=V
        int h   = (nt % 16) >> 1;
        int d   = ((nt & 1) << 4) + ln;
        _Float16* dst = (mat == 0) ? q : ((mat == 1) ? kk : v);
        float sc = (mat == 0) ? QSCALE : 1.0f;

        const size_t hoff   = (size_t)(n * NHD + h);
        const bool   isV    = (mat == 2);
        const size_t base   = isV ? (hoff * HD + d) * (size_t)LSEQ
                                  : hoff * (size_t)LSEQ * HD + d;
        const int    stride = isV ? 1 : HD;

        for (int mt = 0; mt < 4; ++mt) {
            v8f acc = {};
            for (int kt = 0; kt < 8; ++kt) {
                int m = mt * 16 + ln;
                v8h alo = *(const v8h*)&Alds[m * 264 + kt * 32 + 16 * hi];
                v8h ahi = *(const v8h*)&Alds[m * 264 + kt * 32 + 8 + 16 * hi];
                v16h a;
                #pragma unroll
                for (int jj = 0; jj < 8; ++jj) { a[jj] = alo[jj]; a[jj + 8] = ahi[jj]; }
                const v8h* bp = (const v8h*)&wh2[((kt * 48 + nt) * 32 + lane) * 16];
                v8h blo = bp[0], bhi = bp[1];
                v16h b;
                #pragma unroll
                for (int jj = 0; jj < 8; ++jj) { b[jj] = blo[jj]; b[jj + 8] = bhi[jj]; }
                acc = __builtin_amdgcn_wmma_f32_16x16x32_f16(
                        false, a, false, b, (short)0, acc, false, false);
            }
            #pragma unroll
            for (int r = 0; r < 8; ++r) {
                int lq = lbase + mt * 16 + r + 8 * hi;
                dst[base + (size_t)lq * stride] = (_Float16)(acc[r] * sc);
            }
        }
    }
}

// ---------------------------------------------------------------------------
// k2: flash attention. block = 128 q-rows of one (n,h); 8 waves x 16 rows.
// K and V tiles double-buffered in LDS via global_load_async_to_lds_b128;
// prefetch of tile t+1 overlaps the 8 WMMAs + online softmax of tile t.
// ---------------------------------------------------------------------------
__global__ __launch_bounds__(256) void k2_attn(const _Float16* __restrict__ Q,
                                               const _Float16* __restrict__ K,
                                               const _Float16* __restrict__ V,
                                               float* __restrict__ out) {
    __shared__ _Float16 Plds[8][16 * VPAD];      // per-wave P tile (18.0 KB)
    __shared__ _Float16 Kbuf[2][64 * KPAD];      // K tiles, 2 buffers (10.0 KB)
    __shared__ _Float16 Vbuf[2][32 * VPAD];      // V^T tiles, 2 buffers (9.0 KB)
    __shared__ float    Ost[128 * 33];           // output transpose staging (16.5 KB)

    const int qt  = blockIdx.x;                  // 0..31
    const int h   = blockIdx.y;                  // 0..7
    const int n   = blockIdx.z;                  // 0..3
    const int tid = threadIdx.x;
    const int wv = tid >> 5, lane = tid & 31, hi = lane >> 4, ln = lane & 15;

    const size_t headoff = (size_t)(n * NHD + h) * LSEQ * HD;
    const _Float16* Qh  = Q + headoff;
    const _Float16* Kh  = K + headoff;
    const _Float16* Vtg = V + headoff;           // transposed layout [32][LSEQ]
    const int qbase = qt * 128;

    // K-tile async copy: 64 rows x 64B  -> Kbuf rows of 80B
    const int krow = tid >> 2, kch = tid & 3;
    // V-tile async copy: 32 d-rows x 128B -> Vbuf rows of 144B
    const int vd = tid >> 3, vch = tid & 7;

    // prefetch tile 0 into buffer 0
    async_ld16(&Kbuf[0][krow * KPAD + kch * 8], &Kh[(0 + krow) * HD + kch * 8]);
    async_ld16(&Vbuf[0][vd * VPAD + vch * 8], &Vtg[(size_t)vd * LSEQ + 0 + vch * 8]);

    // Q A-fragment (held in registers across the whole loop)
    v16h aq;
    {
        const int lq = qbase + wv * 16 + ln;
        v8h lo  = *(const v8h*)&Qh[lq * HD + 16 * hi];
        v8h hi8 = *(const v8h*)&Qh[lq * HD + 8 + 16 * hi];
        #pragma unroll
        for (int jj = 0; jj < 8; ++jj) { aq[jj] = lo[jj]; aq[jj + 8] = hi8[jj]; }
    }

    v8f o0 = {}, o1 = {};
    float mrow[8], lrow[8];
    #pragma unroll
    for (int r = 0; r < 8; ++r) { mrow[r] = -1e30f; lrow[r] = 0.0f; }

    const int NT = LSEQ / 64;
    for (int kt = 0; kt < NT; ++kt) {
        const int buf = kt & 1;
        __syncthreads();                         // buf^1 readers (tile kt-1) done
        if (kt + 1 < NT) {
            const int kvb2 = (kt + 1) * 64;
            async_ld16(&Kbuf[buf ^ 1][krow * KPAD + kch * 8],
                       &Kh[(kvb2 + krow) * HD + kch * 8]);
            async_ld16(&Vbuf[buf ^ 1][vd * VPAD + vch * 8],
                       &Vtg[(size_t)vd * LSEQ + kvb2 + vch * 8]);
            wait_async_le2();                    // in-order => tile kt's 2 ops done
        } else {
            wait_async_0();
        }
        __syncthreads();                         // tile kt visible to all waves
        const _Float16* Kb = &Kbuf[buf][0];
        const _Float16* Vb = &Vbuf[buf][0];

        // S = q @ k^T  (4 tiles of 16 columns)
        v8f s[4];
        #pragma unroll
        for (int t4 = 0; t4 < 4; ++t4) {
            const v8h* p = (const v8h*)&Kb[(t4 * 16 + ln) * KPAD + 16 * hi];
            v8h lo = p[0], hi8 = p[1];
            v16h bk;
            #pragma unroll
            for (int jj = 0; jj < 8; ++jj) { bk[jj] = lo[jj]; bk[jj + 8] = hi8[jj]; }
            v8f z = {};
            s[t4] = __builtin_amdgcn_wmma_f32_16x16x32_f16(
                        false, aq, false, bk, (short)0, z, false, false);
        }

        // online softmax (row reductions across the 16-lane half-groups)
        #pragma unroll
        for (int r = 0; r < 8; ++r) {
            float tmax = fmaxf(fmaxf(s[0][r], s[1][r]), fmaxf(s[2][r], s[3][r]));
            #pragma unroll
            for (int msk = 1; msk < 16; msk <<= 1)
                tmax = fmaxf(tmax, __shfl_xor(tmax, msk, 32));
            float mnew  = fmaxf(mrow[r], tmax);
            float alpha = __expf(mrow[r] - mnew);
            float rs = 0.0f;
            #pragma unroll
            for (int t4 = 0; t4 < 4; ++t4) {
                float p = __expf(s[t4][r] - mnew);
                s[t4][r] = p;
                rs += p;
            }
            #pragma unroll
            for (int msk = 1; msk < 16; msk <<= 1)
                rs += __shfl_xor(rs, msk, 32);
            lrow[r] = lrow[r] * alpha + rs;
            mrow[r] = mnew;
            o0[r] *= alpha;
            o1[r] *= alpha;
        }

        // P: C/D layout -> A layout through per-wave LDS
        _Float16* pw = &Plds[wv][0];
        #pragma unroll
        for (int t4 = 0; t4 < 4; ++t4)
            #pragma unroll
            for (int r = 0; r < 8; ++r)
                pw[(r + 8 * hi) * VPAD + t4 * 16 + ln] = (_Float16)s[t4][r];

        // O += P @ V
        #pragma unroll
        for (int c = 0; c < 2; ++c) {
            v8h lo  = *(const v8h*)&pw[ln * VPAD + 32 * c + 16 * hi];
            v8h hi8 = *(const v8h*)&pw[ln * VPAD + 32 * c + 8 + 16 * hi];
            v16h ap;
            #pragma unroll
            for (int jj = 0; jj < 8; ++jj) { ap[jj] = lo[jj]; ap[jj + 8] = hi8[jj]; }
            #pragma unroll
            for (int dt = 0; dt < 2; ++dt) {
                const v8h* qv = (const v8h*)&Vb[(dt * 16 + ln) * VPAD + 32 * c + 16 * hi];
                v8h vlo = qv[0], vhi = qv[1];
                v16h bv;
                #pragma unroll
                for (int jj = 0; jj < 8; ++jj) { bv[jj] = vlo[jj]; bv[jj + 8] = vhi[jj]; }
                if (dt == 0)
                    o0 = __builtin_amdgcn_wmma_f32_16x16x32_f16(
                            false, ap, false, bv, (short)0, o0, false, false);
                else
                    o1 = __builtin_amdgcn_wmma_f32_16x16x32_f16(
                            false, ap, false, bv, (short)0, o1, false, false);
            }
        }
    }

    // epilogue: normalize, transpose through LDS, coalesced (N,C,H,W) store
    __syncthreads();
    #pragma unroll
    for (int r = 0; r < 8; ++r) {
        float inv = 1.0f / lrow[r];
        int lr = wv * 16 + r + 8 * hi;
        Ost[lr * 33 + ln]      = o0[r] * inv;
        Ost[lr * 33 + 16 + ln] = o1[r] * inv;
    }
    __syncthreads();
    const size_t outbase = (size_t)(n * CCH + h * HD) * LSEQ + qbase;
    for (int i = tid; i < 128 * 32; i += 256) {
        int d = i >> 7;
        int l = i & 127;
        out[outbase + (size_t)d * LSEQ + l] = Ost[l * 33 + d];
    }
}

extern "C" void kernel_launch(void* const* d_in, const int* in_sizes, int n_in,
                              void* d_out, int out_size, void* d_ws, size_t ws_size,
                              hipStream_t stream) {
    const float* x = (const float*)d_in[0];     // (4,256,64,64) f32
    const float* W = (const float*)d_in[1];     // (256,768) f32
    _Float16* ws  = (_Float16*)d_ws;
    _Float16* wh2 = ws;
    _Float16* q   = ws + Q_OFF;
    _Float16* k   = ws + K_OFF;
    _Float16* v   = ws + V_OFF;
    float* out = (float*)d_out;

    k0_swizzleW<<<768, 256, 0, stream>>>(W, wh2);
    k1_qkv<<<dim3(64, 4), 256, 0, stream>>>(x, wh2, q, k, v);
    k2_attn<<<dim3(32, 8, 4), 256, 0, stream>>>(q, k, v, out);
}
